// AutoEncoder_1119_light_13649406067386
// MI455X (gfx1250) — matrix operand order repaired and checked by reading, not compiled, not run on hardware
//
// MI455X (gfx1250, CDNA5) implementation of the AutoEncoder forward pass.
// Compute-bound (~500 GFLOP vs <1GB traffic @ 23.3 TB/s) -> everything dense
// runs on v_wmma_f32_16x16x32_f16 (f16 in, f32 accum), wave32.
#include <hip/hip_runtime.h>

typedef __attribute__((ext_vector_type(16))) _Float16 v16h;
typedef __attribute__((ext_vector_type(8)))  float    v8f;

union H16 { v16h v; uint4 q[2]; };

#define WMMA(A, B, C) __builtin_amdgcn_wmma_f32_16x16x32_f16(false, (A), false, (B), (short)0, (C), false, false)

static inline int ceil_div(int a, int b) { return (a + b - 1) / b; }

// ---------------------------------------------------------------- utilities
__global__ __launch_bounds__(256) void k_f32h(const float* __restrict__ a, _Float16* __restrict__ o, size_t n) {
  size_t i = (size_t)blockIdx.x * 256 + threadIdx.x;
  if (i < n) o[i] = (_Float16)a[i];
}

__global__ __launch_bounds__(256) void k_zero(float* __restrict__ p, size_t n) {
  size_t i = (size_t)blockIdx.x * 256 + threadIdx.x;
  if (i < n) p[i] = 0.f;
}

__global__ __launch_bounds__(256) void k_fillu32(unsigned* __restrict__ p, unsigned v, size_t n) {
  size_t i = (size_t)blockIdx.x * 256 + threadIdx.x;
  if (i < n) p[i] = v;
}

__global__ __launch_bounds__(256) void k_add_relu(const float* __restrict__ a, const float* __restrict__ b,
                                                  float* __restrict__ o, size_t n) {
  size_t i = (size_t)blockIdx.x * 256 + threadIdx.x;
  if (i < n) { float v = a[i] + b[i]; o[i] = fmaxf(v, 0.f); }
}

// NHWC zero-pad + f32->f16 (pad 1 on W always; pad 1 on H iff padH)
__global__ __launch_bounds__(256) void k_pad(const float* __restrict__ in, _Float16* __restrict__ o,
                                             int F, int H, int W, int C, int padH) {
  int ph = H + 2 * padH, pw = W + 2;
  size_t total = (size_t)F * ph * pw * C;
  size_t i = (size_t)blockIdx.x * 256 + threadIdx.x;
  if (i >= total) return;
  int c = (int)(i % C); size_t r = i / C;
  int xx = (int)(r % pw); r /= pw;
  int yy = (int)(r % ph); int f = (int)(r / ph);
  int x = xx - 1, y = yy - padH;
  float v = 0.f;
  if (x >= 0 && x < W && y >= 0 && y < H) v = in[(((size_t)f * H + y) * W + x) * C + c];
  o[i] = (_Float16)v;
}

// OIHW / OIL conv weight -> f16 [Cout][tap*Cin + ci]
__global__ __launch_bounds__(256) void k_repack(const float* __restrict__ w, _Float16* __restrict__ o,
                                                int Cout, int Cin, int KH, int KW) {
  int total = Cout * Cin * KH * KW;
  int i = blockIdx.x * 256 + threadIdx.x;
  if (i >= total) return;
  int kx = i % KW; int t = i / KW;
  int ky = t % KH; t /= KH;
  int ci = t % Cin; int co = t / Cin;
  o[(size_t)co * (KH * KW * Cin) + (size_t)(ky * KW + kx) * Cin + ci] = (_Float16)w[i];
}

// NHWC [F,S,C] -> [F, C*S] (CHW flatten order)
__global__ __launch_bounds__(256) void k_flatten(const float* __restrict__ in, float* __restrict__ o,
                                                 int F, int S, int C) {
  size_t total = (size_t)F * S * C;
  size_t i = (size_t)blockIdx.x * 256 + threadIdx.x;
  if (i >= total) return;
  int c = (int)(i % C); size_t r = i / C;
  int s = (int)(r % S); int f = (int)(r / S);
  o[(size_t)f * (C * S) + (size_t)c * S + s] = in[i];
}

// NHWC max-pool (2x2 when twod, else 1x2)
__global__ __launch_bounds__(256) void k_pool(const float* __restrict__ in, float* __restrict__ o,
                                              int F, int H, int W, int C, int twod) {
  int H2 = twod ? H / 2 : 1, W2 = W / 2;
  size_t total = (size_t)F * H2 * W2 * C;
  size_t i = (size_t)blockIdx.x * 256 + threadIdx.x;
  if (i >= total) return;
  int c = (int)(i % C); size_t r = i / C;
  int x = (int)(r % W2); r /= W2;
  int y = (int)(r % H2); int f = (int)(r / H2);
  size_t b = (((size_t)f * H + (twod ? y * 2 : 0)) * W + x * 2) * C + c;
  float m = fmaxf(in[b], in[b + C]);
  if (twod) m = fmaxf(m, fmaxf(in[b + (size_t)W * C], in[b + (size_t)W * C + C]));
  o[i] = m;
}

// LayerNorm over D per item; weight index mapped CHW<->NHWC via 'spatial'
__global__ __launch_bounds__(256) void k_ln(const float* __restrict__ in, float* __restrict__ out,
                                            _Float16* __restrict__ o16,
                                            const float* __restrict__ w, const float* __restrict__ b,
                                            int D, int C, int spatial, int act) {
  int item = blockIdx.x;
  const float* p = in + (size_t)item * D;
  float* o = out + (size_t)item * D;
  __shared__ float red[256];
  int tid = threadIdx.x;
  float s = 0.f;
  for (int i = tid; i < D; i += 256) s += p[i];
  red[tid] = s; __syncthreads();
  for (int st = 128; st; st >>= 1) { if (tid < st) red[tid] += red[tid + st]; __syncthreads(); }
  float mean = red[0] / D; __syncthreads();
  float v = 0.f;
  for (int i = tid; i < D; i += 256) { float d0 = p[i] - mean; v += d0 * d0; }
  red[tid] = v; __syncthreads();
  for (int st = 128; st; st >>= 1) { if (tid < st) red[tid] += red[tid + st]; __syncthreads(); }
  float rstd = rsqrtf(red[0] / D + 1e-5f);
  for (int i = tid; i < D; i += 256) {
    int wi = spatial ? ((i % C) * spatial + i / C) : i;
    float y = (p[i] - mean) * rstd * w[wi] + b[wi];
    if (act == 1) y = fmaxf(y, 0.f);
    else if (act == 2) y = y > 0.f ? y : 0.01f * y;
    o[i] = y;
    if (o16) o16[(size_t)item * D + i] = (_Float16)y;
  }
}

// layer-0 convs (Cin=3), direct
__global__ __launch_bounds__(256) void k_conv0_2d(const float* __restrict__ in, const float* __restrict__ w,
                                                  const float* __restrict__ bias, float* __restrict__ out, int F) {
  int total = F * 256 * 32;
  int id = blockIdx.x * 256 + threadIdx.x;
  if (id >= total) return;
  int co = id % 32; int x = (id / 32) % 16; int y = (id / 512) % 16; int f = id / 8192;
  float a = bias[co];
  for (int ci = 0; ci < 3; ci++)
    for (int dy = 0; dy < 3; dy++) {
      int yy = y + dy - 1; if (yy < 0 || yy > 15) continue;
      for (int dx = 0; dx < 3; dx++) {
        int xx = x + dx - 1; if (xx < 0 || xx > 15) continue;
        a += in[(((size_t)f * 3 + ci) * 16 + yy) * 16 + xx] * w[((co * 3 + ci) * 3 + dy) * 3 + dx];
      }
    }
  out[id] = a;
}

__global__ __launch_bounds__(256) void k_conv0_1d(const float* __restrict__ in, const float* __restrict__ w,
                                                  const float* __restrict__ bias, float* __restrict__ out, int F) {
  int total = F * 16 * 32;
  int id = blockIdx.x * 256 + threadIdx.x;
  if (id >= total) return;
  int co = id % 32; int x = (id / 32) % 16; int f = id / 512;
  float a = bias[co];
  for (int ci = 0; ci < 3; ci++)
    for (int dx = 0; dx < 3; dx++) {
      int xx = x + dx - 1; if (xx < 0 || xx > 15) continue;
      a += in[((size_t)f * 3 + ci) * 16 + xx] * w[(co * 3 + ci) * 3 + dx];
    }
  out[id] = a;
}

// ---------------------------------------------------------------- WMMA GEMM
// C[M,N] = act(A[M,K]f16 @ Bt[N,K]f16^T + bias) (+resid); per-wave 32x32 tile.
__global__ __launch_bounds__(256) void k_gemm(const _Float16* __restrict__ A, const _Float16* __restrict__ Bt,
                                              const float* __restrict__ bias, const float* __restrict__ resid,
                                              float* __restrict__ C, _Float16* __restrict__ C16,
                                              int M, int N, int K, int act, int ntn, int tiles) {
  int wid = blockIdx.x * 8 + (threadIdx.x >> 5);
  if (wid >= tiles) return;
  int tm = wid / ntn, tn = wid % ntn;
  int lane = threadIdx.x & 31, m16 = lane & 15, hf = lane >> 4;
  const _Float16* a0 = A + (size_t)(tm * 32 + m16) * K;
  const _Float16* a1 = a0 + (size_t)16 * K;
  const _Float16* b0 = Bt + (size_t)(tn * 32 + m16) * K;
  const _Float16* b1 = b0 + (size_t)16 * K;
  v8f acc[2][2] = {};
  for (int kk = 0; kk < K; kk += 32) {
    H16 af[2], bf[2];
    int sa0 = kk + hf * 16, sa1 = kk + hf * 16 + 8;
    af[0].q[0] = *(const uint4*)(a0 + sa0); af[0].q[1] = *(const uint4*)(a0 + sa1);
    af[1].q[0] = *(const uint4*)(a1 + sa0); af[1].q[1] = *(const uint4*)(a1 + sa1);
    int sb = kk + hf * 16;
    bf[0].q[0] = *(const uint4*)(b0 + sb); bf[0].q[1] = *(const uint4*)(b0 + sb + 8);
    bf[1].q[0] = *(const uint4*)(b1 + sb); bf[1].q[1] = *(const uint4*)(b1 + sb + 8);
    acc[0][0] = WMMA(af[0].v, bf[0].v, acc[0][0]);
    acc[0][1] = WMMA(af[0].v, bf[1].v, acc[0][1]);
    acc[1][0] = WMMA(af[1].v, bf[0].v, acc[1][0]);
    acc[1][1] = WMMA(af[1].v, bf[1].v, acc[1][1]);
  }
#pragma unroll
  for (int i = 0; i < 2; i++)
#pragma unroll
    for (int j = 0; j < 2; j++) {
      int col = tn * 32 + j * 16 + m16;
      float bv = bias ? bias[col] : 0.f;
#pragma unroll
      for (int r = 0; r < 8; r++) {
        int row = tm * 32 + i * 16 + r + 8 * hf;
        float v = acc[i][j][r] + bv;
        if (act == 1) v = fmaxf(v, 0.f);
        else if (act == 2) v = v > 0.f ? v : 0.01f * v;
        size_t o = (size_t)row * N + col;
        if (resid) v += resid[o];
        C[o] = v;
        if (C16) C16[o] = (_Float16)v;
      }
    }
}

// implicit-GEMM conv (KHx3 taps) over zero-padded NHWC f16 input
__global__ __launch_bounds__(256) void k_conv3(const _Float16* __restrict__ P, const _Float16* __restrict__ Wt,
                                               const float* __restrict__ bias, float* __restrict__ C,
                                               int H, int W, int Cin, int Cout, int KH, int Ktot,
                                               int ntn, int tiles) {
  int wid = blockIdx.x * 8 + (threadIdx.x >> 5);
  if (wid >= tiles) return;
  int tm = wid / ntn, tn = wid % ntn;
  int lane = threadIdx.x & 31, m16 = lane & 15, hf = lane >> 4;
  int HW = H * W, pw = W + 2, ph = (KH == 3) ? H + 2 : H;
  size_t abase[2];
#pragma unroll
  for (int g = 0; g < 2; g++) {
    int m = tm * 32 + g * 16 + m16;
    int f = m / HW, rem = m % HW, y = rem / W, x = rem % W;
    abase[g] = (((size_t)f * ph + y) * pw + x) * (size_t)Cin;
  }
  const _Float16* b0 = Wt + (size_t)(tn * 32 + m16) * Ktot;
  const _Float16* b1 = b0 + (size_t)16 * Ktot;
  v8f acc[2][2] = {};
  for (int kk = 0; kk < Ktot; kk += 32) {
    H16 af[2], bf[2];
#pragma unroll
    for (int g = 0; g < 2; g++)
#pragma unroll
      for (int c2 = 0; c2 < 2; c2++) {
        int s = kk + hf * 16 + c2 * 8;
        int t = s / Cin, coff = s - t * Cin;
        int ty = t / 3, tx = t - ty * 3;
        af[g].q[c2] = *(const uint4*)(P + abase[g] + (size_t)(ty * pw + tx) * Cin + coff);
      }
    int sb = kk + hf * 16;
    bf[0].q[0] = *(const uint4*)(b0 + sb); bf[0].q[1] = *(const uint4*)(b0 + sb + 8);
    bf[1].q[0] = *(const uint4*)(b1 + sb); bf[1].q[1] = *(const uint4*)(b1 + sb + 8);
    acc[0][0] = WMMA(af[0].v, bf[0].v, acc[0][0]);
    acc[0][1] = WMMA(af[0].v, bf[1].v, acc[0][1]);
    acc[1][0] = WMMA(af[1].v, bf[0].v, acc[1][0]);
    acc[1][1] = WMMA(af[1].v, bf[1].v, acc[1][1]);
  }
#pragma unroll
  for (int i = 0; i < 2; i++)
#pragma unroll
    for (int j = 0; j < 2; j++) {
      int col = tn * 32 + j * 16 + m16;
      float bv = bias[col];
#pragma unroll
      for (int r = 0; r < 8; r++) {
        int row = tm * 32 + i * 16 + r + 8 * hf;
        C[(size_t)row * Cout + col] = acc[i][j][r] + bv;
      }
    }
}

// ---------------------------------------------------------------- GATv2 bits
__device__ inline unsigned ford(float f) { unsigned u = __float_as_uint(f); return (u & 0x80000000u) ? ~u : (u | 0x80000000u); }
__device__ inline float fdec(unsigned u) { return __uint_as_float((u & 0x80000000u) ? (u & 0x7fffffffu) : ~u); }

__global__ __launch_bounds__(256) void k_count(const int* __restrict__ dst, float* __restrict__ cnt, int E) {
  int e = blockIdx.x * 256 + threadIdx.x;
  if (e < E) atomicAdd(&cnt[dst[e]], 1.f);
}

__global__ __launch_bounds__(256) void k_scatter128(const float* __restrict__ rows, const int* __restrict__ dst,
                                                    float* __restrict__ sums, int E) {
  int e = blockIdx.x * 8 + (threadIdx.x >> 5);
  if (e >= E) return;
  int lane = threadIdx.x & 31, d0 = dst[e];
  for (int d = lane; d < 128; d += 32)
    atomicAdd(&sums[(size_t)d0 * 128 + d], rows[(size_t)e * 128 + d]);
}

__global__ __launch_bounds__(256) void k_mean128(const float* __restrict__ sums, const float* __restrict__ cnt,
                                                 float* __restrict__ mp, int N) {
  int i = blockIdx.x * 256 + threadIdx.x;
  if (i < N * 128) mp[i] = sums[i] / fmaxf(cnt[i >> 7], 1.f);
}

__global__ __launch_bounds__(256) void k_scores(const float* __restrict__ xl, const float* __restrict__ xr,
                                                const float* __restrict__ eproj, const float* __restrict__ mproj,
                                                const float* __restrict__ att, const int* __restrict__ src,
                                                const int* __restrict__ dst, float* __restrict__ sc,
                                                unsigned* __restrict__ nmax, int E, int N) {
  int item = blockIdx.x * 8 + (threadIdx.x >> 5);
  if (item >= E + N) return;
  int lane = threadIdx.x & 31;
  int s, dg; const float* ep;
  if (item < E) { s = src[item]; dg = dst[item]; ep = eproj + (size_t)item * 128; }
  else { s = item - E; dg = s; ep = mproj + (size_t)s * 128; }
  float a = 0.f;
  for (int d = lane; d < 128; d += 32) {
    float m = xl[(size_t)s * 128 + d] + xr[(size_t)dg * 128 + d] + ep[d];
    m = m > 0.f ? m : 0.2f * m;
    a += m * att[d];
  }
  for (int o = 16; o; o >>= 1) a += __shfl_xor(a, o, 32);
  if (lane == 0) { sc[item] = a; atomicMax(&nmax[dg], ford(a)); }
}

__global__ __launch_bounds__(256) void k_expsum(const float* __restrict__ sc, const unsigned* __restrict__ nmax,
                                                const int* __restrict__ dst, float* __restrict__ wexp,
                                                float* __restrict__ esum, int E, int N) {
  int i = blockIdx.x * 256 + threadIdx.x;
  if (i >= E + N) return;
  int seg = (i < E) ? dst[i] : i - E;
  float w = __expf(sc[i] - fdec(nmax[seg]));
  wexp[i] = w;
  atomicAdd(&esum[seg], w);
}

__global__ __launch_bounds__(256) void k_agg(const float* __restrict__ xl, const float* __restrict__ wexp,
                                             const float* __restrict__ esum, const int* __restrict__ src,
                                             const int* __restrict__ dst, float* __restrict__ agg, int E, int N) {
  int item = blockIdx.x * 8 + (threadIdx.x >> 5);
  if (item >= E + N) return;
  int lane = threadIdx.x & 31;
  int seg = (item < E) ? dst[item] : item - E;
  int sr = (item < E) ? src[item] : item - E;
  float alpha = wexp[item] / esum[seg];
  for (int d = lane; d < 128; d += 32)
    atomicAdd(&agg[(size_t)seg * 128 + d], alpha * xl[(size_t)sr * 128 + d]);
}

__global__ __launch_bounds__(256) void k_gatfin(const float* __restrict__ agg, const float* __restrict__ b,
                                                float* __restrict__ x, int N) {
  int i = blockIdx.x * 256 + threadIdx.x;
  if (i >= N * 128) return;
  float v = agg[i] + b[i & 127];
  x[i] = v > 0.f ? v : 0.01f * v;
}

// ---------------------------------------------------------------- attention
__global__ __launch_bounds__(256) void k_qkvprep(const float* __restrict__ qkv, _Float16* __restrict__ Qp,
                                                 _Float16* __restrict__ Kp, _Float16* __restrict__ Vt) {
  int id = blockIdx.x * 256 + threadIdx.x;
  if (id >= 16 * 2048 * 64) return;
  int d = id & 63, s = (id >> 6) & 2047, h = id >> 17;
  const float scl = 0.14433756729740643f;  // 1/sqrt(48)
  float q = 0.f, k = 0.f, v = 0.f;
  if (d < 48) {
    size_t b = (size_t)s * 2304 + h * 48 + d;
    q = qkv[b] * scl; k = qkv[b + 768]; v = qkv[b + 1536];
  }
  Qp[id] = (_Float16)q;
  Kp[id] = (_Float16)k;
  Vt[((size_t)h * 64 + d) * 2048 + s] = (_Float16)v;
}

// Flash attention: 32 queries/block, head-dim 48 padded to 64, WMMA for QK^T and P.V
__global__ __launch_bounds__(256) void k_flash(const _Float16* __restrict__ Qp, const _Float16* __restrict__ Kp,
                                               const _Float16* __restrict__ Vt, float* __restrict__ Out) {
  const int S = 2048, HD = 64;
  int head = blockIdx.x >> 6;
  int qbase = (blockIdx.x & 63) * 32;
  int tid = threadIdx.x, lane = tid & 31, w = tid >> 5;
  int mi = w >> 2, ni = w & 3;
  int m16 = lane & 15, hf = lane >> 4;

  __shared__ __align__(16) _Float16 Qs[32][64];
  __shared__ __align__(16) float    Ss[32][64];
  __shared__ __align__(16) _Float16 Ps[32][64];
  __shared__ float mrow[32], lrow[32], crow[32];

  const _Float16* Qg = Qp + ((size_t)head * S + qbase) * HD;
  for (int i = tid; i < 32 * 64; i += 256) Qs[i >> 6][i & 63] = Qg[i];
  if (tid < 32) { mrow[tid] = -__builtin_inff(); lrow[tid] = 0.f; }
  __syncthreads();

  H16 qa0, qa1;
  {
    const _Float16* qr = &Qs[mi * 16 + m16][0];
    qa0.q[0] = *(const uint4*)(qr + hf * 16);      qa0.q[1] = *(const uint4*)(qr + hf * 16 + 8);
    qa1.q[0] = *(const uint4*)(qr + 32 + hf * 16); qa1.q[1] = *(const uint4*)(qr + 32 + hf * 16 + 8);
  }
  v8f acc = {};
  const _Float16* Kh = Kp + (size_t)head * S * HD;
  const _Float16* Vh = Vt + (size_t)head * HD * S;

  for (int kc = 0; kc < S; kc += 64) {
    {  // S tile
      const _Float16* kr = Kh + (size_t)(kc + ni * 16 + m16) * HD;
      H16 kb0, kb1;
      kb0.q[0] = *(const uint4*)(kr + hf * 16);      kb0.q[1] = *(const uint4*)(kr + hf * 16 + 8);
      kb1.q[0] = *(const uint4*)(kr + 32 + hf * 16); kb1.q[1] = *(const uint4*)(kr + 32 + hf * 16 + 8);
      v8f st = {};
      st = WMMA(qa0.v, kb0.v, st);
      st = WMMA(qa1.v, kb1.v, st);
#pragma unroll
      for (int r = 0; r < 8; r++) Ss[mi * 16 + r + 8 * hf][ni * 16 + m16] = st[r];
    }
    __syncthreads();
    {  // online softmax (8 threads per row)
      int row = tid >> 3, j0 = (tid & 7) << 3;
      float sv[8], mx = -__builtin_inff();
#pragma unroll
      for (int j = 0; j < 8; j++) { sv[j] = Ss[row][j0 + j]; mx = fmaxf(mx, sv[j]); }
      for (int o = 4; o; o >>= 1) mx = fmaxf(mx, __shfl_xor(mx, o, 8));
      float mold = mrow[row];
      float mnew = fmaxf(mold, mx);
      float ps = 0.f;
#pragma unroll
      for (int j = 0; j < 8; j++) { float e = __expf(sv[j] - mnew); Ps[row][j0 + j] = (_Float16)e; ps += e; }
      for (int o = 4; o; o >>= 1) ps += __shfl_xor(ps, o, 8);
      if ((tid & 7) == 0) { float c = __expf(mold - mnew); crow[row] = c; lrow[row] = lrow[row] * c + ps; mrow[row] = mnew; }
    }
    __syncthreads();
    {  // rescale accumulator + P @ V
#pragma unroll
      for (int r = 0; r < 8; r++) acc[r] *= crow[mi * 16 + r + 8 * hf];
      const _Float16* pr = &Ps[mi * 16 + m16][0];
      H16 pa0, pa1, vb0, vb1;
      pa0.q[0] = *(const uint4*)(pr + hf * 16);      pa0.q[1] = *(const uint4*)(pr + hf * 16 + 8);
      pa1.q[0] = *(const uint4*)(pr + 32 + hf * 16); pa1.q[1] = *(const uint4*)(pr + 32 + hf * 16 + 8);
      const _Float16* vr = Vh + (size_t)(ni * 16 + m16) * S + kc;
      vb0.q[0] = *(const uint4*)(vr + hf * 16);      vb0.q[1] = *(const uint4*)(vr + hf * 16 + 8);
      vb1.q[0] = *(const uint4*)(vr + 32 + hf * 16); vb1.q[1] = *(const uint4*)(vr + 32 + hf * 16 + 8);
      acc = WMMA(pa0.v, vb0.v, acc);
      acc = WMMA(pa1.v, vb1.v, acc);
    }
    __syncthreads();
  }
  if (ni < 3) {  // dims 48..63 are padding
#pragma unroll
    for (int r = 0; r < 8; r++) {
      int row = mi * 16 + r + 8 * hf;
      Out[(size_t)(qbase + row) * 768 + head * 48 + ni * 16 + m16] = acc[r] / lrow[row];
    }
  }
}

// ---------------------------------------------------------------- host side
struct RBp { const float *c1w, *c1b, *n1w, *n1b, *c2w, *c2b, *n2w, *n2b, *dsw, *dnw, *dnb; int ds; };
struct CNNp { const float *c0w, *c0b, *n0w, *n0b; RBp rb[4]; const float *cfw, *cfb; };
struct GATpr { const float *lw, *lb, *rw, *rb_, *ew, *att, *b; };
struct TLp { const float *qkv_w, *qkv_b, *out_w, *out_b, *ln1w, *ln1b, *ln2w, *ln2b, *f1w, *f1b, *f2w, *f2b; };

static void gemm16(hipStream_t st, const _Float16* A, const float* Wf, _Float16* wb16,
                   const float* bias, const float* resid, float* C, _Float16* C16,
                   int M, int N, int K, int act) {
  size_t nk = (size_t)N * K;
  k_f32h<<<ceil_div((int)nk, 256), 256, 0, st>>>(Wf, wb16, nk);
  int ntn = N / 32, tiles = (M / 32) * ntn;
  k_gemm<<<ceil_div(tiles, 8), 256, 0, st>>>(A, wb16, bias, resid, C, C16, M, N, K, act, ntn, tiles);
}

static void conv3run(hipStream_t st, const float* act, _Float16* padbuf, const float* w, _Float16* wb16,
                     const float* bias, float* out, int F, int H, int W, int Cin, int Cout, int KH) {
  int padH = (KH == 3) ? 1 : 0;
  int ph = H + 2 * padH, pw = W + 2;
  size_t pn = (size_t)F * ph * pw * Cin;
  k_pad<<<ceil_div((int)pn, 256), 256, 0, st>>>(act, padbuf, F, H, W, Cin, padH);
  k_repack<<<ceil_div(Cout * Cin * KH * 3, 256), 256, 0, st>>>(w, wb16, Cout, Cin, KH, 3);
  int Ktot = KH * 3 * Cin, M = F * H * W;
  int ntn = Cout / 32, tiles = (M / 32) * ntn;
  k_conv3<<<ceil_div(tiles, 8), 256, 0, st>>>(padbuf, wb16, bias, out, H, W, Cin, Cout, KH, Ktot, ntn, tiles);
}

static void lnrun(hipStream_t st, const float* in, float* out, _Float16* o16,
                  const float* w, const float* b, int items, int D, int C, int spatial, int act) {
  k_ln<<<items, 256, 0, st>>>(in, out, o16, w, b, D, C, spatial, act);
}

// out may alias u0
static void resblock(hipStream_t st, const RBp& p, const float* xin, float* out,
                     float* u0, float* u1, float* u2, _Float16* padbuf, _Float16* wb16, _Float16* a16,
                     int F, int H, int W, int Cin, int Cout, int KH) {
  int sp = H * W;
  conv3run(st, xin, padbuf, p.c1w, wb16, p.c1b, u0, F, H, W, Cin, Cout, KH);
  lnrun(st, u0, u1, nullptr, p.n1w, p.n1b, F, sp * Cout, Cout, sp, 1);
  conv3run(st, u1, padbuf, p.c2w, wb16, p.c2b, u0, F, H, W, Cout, Cout, KH);
  lnrun(st, u0, u2, nullptr, p.n2w, p.n2b, F, sp * Cout, Cout, sp, 0);
  const float* idn = xin;
  if (p.ds) {
    size_t n = (size_t)F * sp * Cin;
    k_f32h<<<ceil_div((int)n, 256), 256, 0, st>>>(xin, a16, n);
    gemm16(st, a16, p.dsw, wb16, nullptr, nullptr, u0, nullptr, F * sp, Cout, Cin, 0);
    lnrun(st, u0, u1, nullptr, p.dnw, p.dnb, F, sp * Cout, Cout, sp, 0);
    idn = u1;
  }
  size_t n = (size_t)F * sp * Cout;
  k_add_relu<<<ceil_div((int)n, 256), 256, 0, st>>>(u2, idn, out, n);
}

extern "C" void kernel_launch(void* const* d_in, const int* in_sizes, int n_in,
                              void* d_out, int out_size, void* d_ws, size_t ws_size,
                              hipStream_t stream) {
  (void)in_sizes; (void)n_in; (void)out_size;
  const int NF = 2048, NE = 16384;
  int ip = 0;
  auto F32 = [&]() { return (const float*)d_in[ip++]; };

  const float* face_pts = F32();
  const float* edge_pts = F32();
  const int* eidx = (const int*)d_in[ip++];
  const int* srcI = eidx;
  const int* dstI = eidx + NE;

  auto readRB = [&](int ds) {
    RBp r{};
    r.c1w = F32(); r.c1b = F32(); r.n1w = F32(); r.n1b = F32();
    r.c2w = F32(); r.c2b = F32(); r.n2w = F32(); r.n2b = F32();
    r.ds = ds;
    if (ds) { r.dsw = F32(); r.dnw = F32(); r.dnb = F32(); }
    return r;
  };
  auto readCNN = [&]() {
    CNNp c{};
    c.c0w = F32(); c.c0b = F32(); c.n0w = F32(); c.n0b = F32();
    c.rb[0] = readRB(1); c.rb[1] = readRB(1); c.rb[2] = readRB(1); c.rb[3] = readRB(0);
    c.cfw = F32(); c.cfb = F32();
    return c;
  };
  CNNp face = readCNN();
  CNNp edge = readCNN();
  GATpr gat[5];
  for (int i = 0; i < 5; i++) {
    gat[i].lw = F32(); gat[i].lb = F32(); gat[i].rw = F32(); gat[i].rb_ = F32();
    gat[i].ew = F32(); gat[i].att = F32(); gat[i].b = F32();
  }
  const float *pin_w = F32(), *pin_b = F32();
  TLp tl[8];
  for (int i = 0; i < 8; i++) {
    tl[i].qkv_w = F32(); tl[i].qkv_b = F32(); tl[i].out_w = F32(); tl[i].out_b = F32();
    tl[i].ln1w = F32(); tl[i].ln1b = F32(); tl[i].ln2w = F32(); tl[i].ln2b = F32();
    tl[i].f1w = F32(); tl[i].f1b = F32(); tl[i].f2w = F32(); tl[i].f2b = F32();
  }
  const float *flnw = F32(), *flnb = F32(), *pout_w = F32(), *pout_b = F32();

  // ---- workspace carve ----
  char* base = (char*)d_ws;
  size_t off = 0;
  auto A32 = [&](size_t n) { float* p = (float*)(base + off); off = (off + n * 4 + 255) & ~(size_t)255; return p; };
  auto A16b = [&](size_t n) { _Float16* p = (_Float16*)(base + off); off = (off + n * 2 + 255) & ~(size_t)255; return p; };
  auto AU32 = [&](size_t n) { unsigned* p = (unsigned*)(base + off); off = (off + n * 4 + 255) & ~(size_t)255; return p; };

  float* xbuf = A32((size_t)NF * 128);
  float* xl = A32((size_t)NF * 128);
  float* xr = A32((size_t)NF * 128);
  float* eproj = A32((size_t)NE * 128);
  float* msum = A32((size_t)NF * 128);
  float* mproj = A32((size_t)NF * 128);
  float* agg = A32((size_t)NF * 128);
  float* cnt = A32(NF);
  float* esum = A32(NF);
  float* sc = A32(NE + NF);
  float* wexp = A32(NE + NF);
  unsigned* nmax = AU32(NF);
  float* ea = A32((size_t)NE * 256);
  float* ybuf = A32((size_t)NF * 768);
  float* tln = A32((size_t)NF * 768);
  float* qkvb = A32((size_t)NF * 2304);
  float* attno = A32((size_t)NF * 768);
  float* ffn1 = A32((size_t)NF * 2048);
  float* sA = A32(2097152);
  float* sB = A32(2097152);
  float* sC = A32(2097152);
  float* sD = A32(2097152);
  _Float16* ea16 = A16b((size_t)NE * 256);
  _Float16* act16 = A16b((size_t)NF * 768);
  _Float16* ln16 = A16b((size_t)NF * 768);
  _Float16* ffn16 = A16b((size_t)NF * 2048);
  _Float16* x16p = A16b((size_t)NF * 128);
  _Float16* wb16 = A16b((size_t)2304 * 768);
  _Float16* padbuf = A16b((size_t)128 * 18 * 18 * 64);
  _Float16* Qp = A16b((size_t)16 * 2048 * 64);
  _Float16* Kp = A16b((size_t)16 * 2048 * 64);
  _Float16* Vt = A16b((size_t)16 * 2048 * 64);
  if (off > ws_size) return;  // not enough scratch: bail rather than corrupt

  // ================= face CNN (chunks of 128 faces) =================
  const int FC = 128;
  for (int ch = 0; ch < NF / FC; ch++) {
    const float* in = face_pts + (size_t)ch * FC * 3 * 256;
    k_conv0_2d<<<ceil_div(FC * 256 * 32, 256), 256, 0, stream>>>(in, face.c0w, face.c0b, sB, FC);
    lnrun(stream, sB, sA, nullptr, face.n0w, face.n0b, FC, 256 * 32, 32, 256, 2);
    resblock(stream, face.rb[0], sA, sB, sB, sC, sD, padbuf, wb16, act16, FC, 16, 16, 32, 64, 3);
    k_pool<<<ceil_div(FC * 64 * 64, 256), 256, 0, stream>>>(sB, sA, FC, 16, 16, 64, 1);
    resblock(stream, face.rb[1], sA, sB, sB, sC, sD, padbuf, wb16, act16, FC, 8, 8, 64, 128, 3);
    k_pool<<<ceil_div(FC * 16 * 128, 256), 256, 0, stream>>>(sB, sA, FC, 8, 8, 128, 1);
    resblock(stream, face.rb[2], sA, sB, sB, sC, sD, padbuf, wb16, act16, FC, 4, 4, 128, 256, 3);
    k_pool<<<ceil_div(FC * 4 * 256, 256), 256, 0, stream>>>(sB, sA, FC, 4, 4, 256, 1);
    resblock(stream, face.rb[3], sA, sB, sB, sC, sD, padbuf, wb16, act16, FC, 2, 2, 256, 256, 3);
    size_t n = (size_t)FC * 4 * 256;
    k_f32h<<<ceil_div((int)n, 256), 256, 0, stream>>>(sB, act16, n);
    gemm16(stream, act16, face.cfw, wb16, face.cfb, nullptr, sC, nullptr, FC * 4, 32, 256, 0);
    k_flatten<<<ceil_div(FC * 4 * 32, 256), 256, 0, stream>>>(sC, xbuf + (size_t)ch * FC * 128, FC, 4, 32);
  }

  // ================= edge CNN (chunks of 1024 edges) =================
  const int EC = 1024;
  for (int ch = 0; ch < NE / EC; ch++) {
    const float* in = edge_pts + (size_t)ch * EC * 3 * 16;
    k_conv0_1d<<<ceil_div(EC * 16 * 32, 256), 256, 0, stream>>>(in, edge.c0w, edge.c0b, sB, EC);
    lnrun(stream, sB, sA, nullptr, edge.n0w, edge.n0b, EC, 16 * 32, 32, 16, 2);
    resblock(stream, edge.rb[0], sA, sB, sB, sC, sD, padbuf, wb16, act16, EC, 1, 16, 32, 64, 1);
    k_pool<<<ceil_div(EC * 8 * 64, 256), 256, 0, stream>>>(sB, sA, EC, 1, 16, 64, 0);
    resblock(stream, edge.rb[1], sA, sB, sB, sC, sD, padbuf, wb16, act16, EC, 1, 8, 64, 128, 1);
    k_pool<<<ceil_div(EC * 4 * 128, 256), 256, 0, stream>>>(sB, sA, EC, 1, 8, 128, 0);
    resblock(stream, edge.rb[2], sA, sB, sB, sC, sD, padbuf, wb16, act16, EC, 1, 4, 128, 256, 1);
    k_pool<<<ceil_div(EC * 2 * 256, 256), 256, 0, stream>>>(sB, sA, EC, 1, 4, 256, 0);
    resblock(stream, edge.rb[3], sA, sB, sB, sC, sD, padbuf, wb16, act16, EC, 1, 2, 256, 256, 1);
    size_t n = (size_t)EC * 2 * 256;
    k_f32h<<<ceil_div((int)n, 256), 256, 0, stream>>>(sB, act16, n);
    gemm16(stream, act16, edge.cfw, wb16, edge.cfb, nullptr, sC, nullptr, EC * 2, 128, 256, 0);
    k_flatten<<<ceil_div(EC * 2 * 128, 256), 256, 0, stream>>>(sC, ea + (size_t)ch * EC * 256, EC, 2, 128);
  }

  // ================= GATv2 x5 =================
  k_f32h<<<ceil_div(NE * 256, 256), 256, 0, stream>>>(ea, ea16, (size_t)NE * 256);
  k_zero<<<ceil_div(NF, 256), 256, 0, stream>>>(cnt, NF);
  k_count<<<ceil_div(NE, 256), 256, 0, stream>>>(dstI, cnt, NE);
  for (int l = 0; l < 5; l++) {
    k_f32h<<<ceil_div(NF * 128, 256), 256, 0, stream>>>(xbuf, x16p, (size_t)NF * 128);
    gemm16(stream, x16p, gat[l].lw, wb16, gat[l].lb, nullptr, xl, nullptr, NF, 128, 128, 0);
    gemm16(stream, x16p, gat[l].rw, wb16, gat[l].rb_, nullptr, xr, nullptr, NF, 128, 128, 0);
    gemm16(stream, ea16, gat[l].ew, wb16, nullptr, nullptr, eproj, nullptr, NE, 128, 256, 0);
    k_zero<<<ceil_div(NF * 128, 256), 256, 0, stream>>>(msum, (size_t)NF * 128);
    k_zero<<<ceil_div(NF * 128, 256), 256, 0, stream>>>(agg, (size_t)NF * 128);
    k_zero<<<ceil_div(NF, 256), 256, 0, stream>>>(esum, NF);
    k_fillu32<<<ceil_div(NF, 256), 256, 0, stream>>>(nmax, 0x007FFFFFu, NF);  // ford(-inf)
    k_scatter128<<<ceil_div(NE, 8), 256, 0, stream>>>(eproj, dstI, msum, NE);
    k_mean128<<<ceil_div(NF * 128, 256), 256, 0, stream>>>(msum, cnt, mproj, NF);
    k_scores<<<ceil_div(NE + NF, 8), 256, 0, stream>>>(xl, xr, eproj, mproj, gat[l].att, srcI, dstI, sc, nmax, NE, NF);
    k_expsum<<<ceil_div(NE + NF, 256), 256, 0, stream>>>(sc, nmax, dstI, wexp, esum, NE, NF);
    k_agg<<<ceil_div(NE + NF, 8), 256, 0, stream>>>(xl, wexp, esum, srcI, dstI, agg, NE, NF);
    k_gatfin<<<ceil_div(NF * 128, 256), 256, 0, stream>>>(agg, gat[l].b, xbuf, NF);
  }

  // ================= transformer fuser =================
  k_f32h<<<ceil_div(NF * 128, 256), 256, 0, stream>>>(xbuf, x16p, (size_t)NF * 128);
  gemm16(stream, x16p, pin_w, wb16, pin_b, nullptr, ybuf, nullptr, NF, 768, 128, 0);
  for (int l = 0; l < 8; l++) {
    lnrun(stream, ybuf, tln, ln16, tl[l].ln1w, tl[l].ln1b, NF, 768, 1, 0, 0);
    gemm16(stream, ln16, tl[l].qkv_w, wb16, tl[l].qkv_b, nullptr, qkvb, nullptr, NF, 2304, 768, 0);
    k_qkvprep<<<ceil_div(16 * 2048 * 64, 256), 256, 0, stream>>>(qkvb, Qp, Kp, Vt);
    k_flash<<<16 * 64, 256, 0, stream>>>(Qp, Kp, Vt, attno);
    k_f32h<<<ceil_div(NF * 768, 256), 256, 0, stream>>>(attno, act16, (size_t)NF * 768);
    gemm16(stream, act16, tl[l].out_w, wb16, tl[l].out_b, ybuf, ybuf, nullptr, NF, 768, 768, 0);
    lnrun(stream, ybuf, tln, ln16, tl[l].ln2w, tl[l].ln2b, NF, 768, 1, 0, 0);
    gemm16(stream, ln16, tl[l].f1w, wb16, tl[l].f1b, nullptr, ffn1, ffn16, NF, 2048, 768, 1);
    gemm16(stream, ffn16, tl[l].f2w, wb16, tl[l].f2b, ybuf, ybuf, nullptr, NF, 768, 2048, 0);
  }
  lnrun(stream, ybuf, tln, ln16, flnw, flnb, NF, 768, 1, 0, 0);
  gemm16(stream, ln16, pout_w, wb16, pout_b, xbuf, (float*)d_out, nullptr, NF, 128, 768, 0);
}
